// AttnBlock1d_61787399520366
// MI455X (gfx1250) — matrix operand order repaired and checked by reading, not compiled, not run on hardware
//
#include <hip/hip_runtime.h>

namespace {
constexpr int B = 8, C = 256, L = 4096, G = 16;
constexpr float EPS = 1e-5f;
constexpr float SCALE = 0.0625f; // C^-0.5 = 1/16
}

typedef __attribute__((ext_vector_type(16))) _Float16 v16h;
typedef __attribute__((ext_vector_type(8)))  _Float16 v8h;
typedef __attribute__((ext_vector_type(8)))  float    v8f;

// A-fragment (16x32 f16, MxK): lane row = lane&15, kb = (lane>>4)*16.
// p must point at (row base + kb); elements [0..7] from p, [8..15] from p+8.
__device__ __forceinline__ v16h frag_a(const _Float16* p) {
  v8h lo = *(const v8h*)(p);
  v8h hi = *(const v8h*)(p + 8);
  v16h r;
#pragma unroll
  for (int t = 0; t < 8; ++t) { r[t] = lo[t]; r[t + 8] = hi[t]; }
  return r;
}

// B-fragment (32x16 f16, KxN): lane col = lane&15, kb = (lane>>4)*16.
// p points at 16 consecutive K values for this lane's column.
__device__ __forceinline__ v16h frag_b(const _Float16* p) {
  v8h lo = *(const v8h*)(p);
  v8h hi = *(const v8h*)(p + 8);
  v16h r;
#pragma unroll
  for (int t = 0; t < 8; ++t) { r[t] = lo[t]; r[t + 8] = hi[t]; }
  return r;
}

// B-fragment built from 16 consecutive f32 (on-the-fly weight conversion).
__device__ __forceinline__ v16h frag_bf(const float* p) {
  v16h r;
#pragma unroll
  for (int t = 0; t < 16; ++t) r[t] = (_Float16)p[t];
  return r;
}

__device__ __forceinline__ v8f wmma_f16(v16h a, v16h b, v8f c) {
  return __builtin_amdgcn_wmma_f32_16x16x32_f16(false, a, false, b, (short)0, c,
                                                false, false);
}

// Async global->LDS copy of one 16-byte chunk per lane (ASYNCcnt-tracked).
__device__ __forceinline__ void async_b128(uint32_t lds_addr, uint64_t gaddr) {
  asm volatile("global_load_async_to_lds_b128 %0, %1, off"
               :: "v"(lds_addr), "v"(gaddr) : "memory");
}

// ---------------------------------------------------------------------------
// Kernel 1: GroupNorm over (C/G, L) per (b,g); writes h transposed ht[b][l][c]
// as f16 (c contiguous) so it is directly WMMA-A-fragment friendly.
// ---------------------------------------------------------------------------
__global__ void gn_kernel(const float* __restrict__ x, const float* __restrict__ w,
                          const float* __restrict__ bgn, _Float16* __restrict__ ht) {
  const int b = blockIdx.y, g = blockIdx.x, tid = threadIdx.x;
  const int CPG = C / G;              // 16
  const int N = CPG * L;              // 65536
  const float* xb = x + ((size_t)b * C + (size_t)g * CPG) * L;
  float s = 0.f, ss = 0.f;
  for (int i = tid; i < N; i += 256) { float v = xb[i]; s += v; ss += v * v; }
  __shared__ float r0[256], r1[256];
  r0[tid] = s; r1[tid] = ss; __syncthreads();
  for (int off = 128; off > 0; off >>= 1) {
    if (tid < off) { r0[tid] += r0[tid + off]; r1[tid] += r1[tid + off]; }
    __syncthreads();
  }
  const float mean = r0[0] * (1.f / N);
  const float var  = r1[0] * (1.f / N) - mean * mean;
  const float rs   = rsqrtf(var + EPS);
  for (int i = tid; i < N; i += 256) {
    int cl = i >> 12;                 // i / L
    int l  = i & (L - 1);
    int ch = g * CPG + cl;
    float v = (xb[i] - mean) * rs * w[ch] + bgn[ch];
    ht[((size_t)b * L + l) * C + ch] = (_Float16)v;
  }
}

// ---------------------------------------------------------------------------
// Kernel 2: QKV projections. One wave = one 16x16 output tile, K=256 in 8 WMMAs.
// z=0 -> Qt[b][l][c] (transposed), z=1 -> Kt[b][j][c], z=2 -> V[b][c][l] natural.
// ---------------------------------------------------------------------------
__global__ void qkv_kernel(const _Float16* __restrict__ ht,
                           const float* __restrict__ wq, const float* __restrict__ bq,
                           const float* __restrict__ wk, const float* __restrict__ bk,
                           const float* __restrict__ wv, const float* __restrict__ bv,
                           _Float16* __restrict__ qt, _Float16* __restrict__ kt,
                           _Float16* __restrict__ vv) {
  const int wave = threadIdx.x >> 5, lane = threadIdx.x & 31;
  const int b = blockIdx.y, z = blockIdx.z;
  const int tt = blockIdx.x * 8 + wave;          // 0..4095
  const int l0 = (tt >> 4) * 16, o0 = (tt & 15) * 16;
  const int n = lane & 15, hi = lane >> 4;
  const int kbA = hi * 16, kbB = hi * 16;
  const float* W    = (z == 0) ? wq : ((z == 1) ? wk : wv);
  const float* bias = (z == 0) ? bq : ((z == 1) ? bk : bv);

  const _Float16* arow = ht + ((size_t)b * L + l0 + n) * C + kbA;
  const float*    brow = W + (size_t)(o0 + n) * C + kbB;

  v8f acc = {};
#pragma unroll
  for (int cb = 0; cb < 8; ++cb)
    acc = wmma_f16(frag_a(arow + cb * 32), frag_bf(brow + cb * 32), acc);

  const float bb = bias[o0 + n];
#pragma unroll
  for (int r = 0; r < 8; ++r) acc[r] += bb;

  if (z < 2) {                                    // store transposed (l, c)
    _Float16* dst = (z == 0) ? qt : kt;
#pragma unroll
    for (int r = 0; r < 8; ++r)
      dst[((size_t)b * L + l0 + r + 8 * hi) * C + o0 + n] = (_Float16)acc[r];
  } else {                                        // V natural (c, l): contiguous 8 halves
    v8h ph;
#pragma unroll
    for (int r = 0; r < 8; ++r) ph[r] = (_Float16)acc[r];
    *(v8h*)(vv + ((size_t)b * C + o0 + n) * L + l0 + 8 * hi) = ph;
  }
}

// ---------------------------------------------------------------------------
// Kernel 3: flash attention with async global->LDS staging of K/V tiles.
// One wave = 16 queries, full 256-channel f32 output accumulator (16 tiles).
// All 8 waves of a block share one batch and the same j-tiles, so each 32-key
// step stages K (16 KB, contiguous) and V (16 KB, strided) into LDS ONCE per
// block via global_load_async_to_lds_b128 (8x less L2 traffic than per-wave
// global fragment reads), then all fragment B-loads come from LDS.
// ---------------------------------------------------------------------------
__global__ void attn_kernel(const _Float16* __restrict__ qt,
                            const _Float16* __restrict__ kt,
                            const _Float16* __restrict__ vv,
                            _Float16* __restrict__ ot) {
  __shared__ __align__(16) _Float16 kbuf[32 * 256];   // (j_local, c) rows of 512B
  __shared__ __align__(16) _Float16 vbuf[256 * 32];   // (c, j_local) rows of 64B
  __shared__ __align__(32) _Float16 plds[8][16 * 32];

  const int wave = threadIdx.x >> 5, lane = threadIdx.x & 31;
  const int gw = blockIdx.x * 8 + wave;          // 0..2047
  const int b = blockIdx.x >> 5;                 // 32 blocks per batch
  const int i0 = (gw & 255) * 16;
  const int n = lane & 15, hi = lane >> 4;
  const int kbA = hi * 16, kbB = hi * 16;
  _Float16* myp = &plds[wave][0];

  const uint32_t kbuf_lds = (uint32_t)(size_t)(void*)kbuf;
  const uint32_t vbuf_lds = (uint32_t)(size_t)(void*)vbuf;
  const char* kt_b = (const char*)(kt + (size_t)b * L * C);
  const char* vv_b = (const char*)(vv + (size_t)b * C * L);

  // Preload all 8 Q A-fragments (K=256) for this 16-query tile.
  v16h qf[8];
  const _Float16* qrow = qt + ((size_t)b * L + i0 + n) * C + kbA;
#pragma unroll
  for (int cb = 0; cb < 8; ++cb) qf[cb] = frag_a(qrow + cb * 32);

  v8f o[16];
#pragma unroll
  for (int t = 0; t < 16; ++t) o[t] = (v8f){};
  float m[8], lsum[8];
#pragma unroll
  for (int r = 0; r < 8; ++r) { m[r] = -1e30f; lsum[r] = 0.f; }

  const int id0 = (wave & 3) * 32 + lane;        // 0..127 within 4-wave group

  for (int j0 = 0; j0 < L; j0 += 32) {
    // ---- stage K/V tiles for this j-step into LDS (ASYNCcnt path) ----
    if (wave < 4) {
      // K tile: 16 KB contiguous at kt_b + j0*512
#pragma unroll
      for (int q = 0; q < 8; ++q) {
        int id = id0 + 128 * q;                  // 0..1023 sixteen-byte chunks
        async_b128(kbuf_lds + id * 16,
                   (uint64_t)(kt_b + (size_t)j0 * 512 + (size_t)id * 16));
      }
    } else {
      // V tile: 256 rows (c) x 64 B, row stride L*2 bytes
#pragma unroll
      for (int q = 0; q < 8; ++q) {
        int id = id0 + 128 * q;                  // 0..1023
        int row = id >> 2, chunk = id & 3;
        async_b128(vbuf_lds + id * 16,
                   (uint64_t)(vv_b + (size_t)row * (L * 2) +
                              (size_t)j0 * 2 + (size_t)chunk * 16));
      }
    }
    asm volatile("s_wait_asynccnt 0x0" ::: "memory");  // my copies arrived
    __syncthreads();                                   // everyone's copies arrived

    // ---- scores: two 16x16 S tiles from LDS K, reduction over C in 8 steps ----
    v8f s0 = {}, s1 = {};
#pragma unroll
    for (int cb = 0; cb < 8; ++cb) {
      v16h bk0 = frag_b(&kbuf[(size_t)(0 + n) * C + cb * 32 + kbB]);
      v16h bk1 = frag_b(&kbuf[(size_t)(16 + n) * C + cb * 32 + kbB]);
      s0 = wmma_f16(qf[cb], bk0, s0);
      s1 = wmma_f16(qf[cb], bk1, s1);
    }
    // ---- online softmax (row = r + 8*hi lives in one 16-lane half) ----
#pragma unroll
    for (int r = 0; r < 8; ++r) {
      float v0 = s0[r] * SCALE;
      float v1 = s1[r] * SCALE;
      float mt = fmaxf(v0, v1);
#pragma unroll
      for (int off = 1; off < 16; off <<= 1) mt = fmaxf(mt, __shfl_xor(mt, off, 32));
      float mn = fmaxf(m[r], mt);
      float corr = __expf(m[r] - mn);
      float p0 = __expf(v0 - mn), p1 = __expf(v1 - mn);
      float rsm = p0 + p1;
#pragma unroll
      for (int off = 1; off < 16; off <<= 1) rsm += __shfl_xor(rsm, off, 32);
      lsum[r] = lsum[r] * corr + rsm;
      m[r] = mn;
      const int ri = r + 8 * hi;
      myp[ri * 32 + n]      = (_Float16)p0;
      myp[ri * 32 + 16 + n] = (_Float16)p1;
#pragma unroll
      for (int t = 0; t < 16; ++t) o[t][r] *= corr;    // rescale O rows
    }
    asm volatile("s_wait_dscnt 0x0" ::: "memory");     // LDS store -> load, same wave
    // ---- O += P (16x32) x V^T (32x16 per c-tile) from LDS V ----
    v16h pa = frag_a(myp + n * 32 + kbA);
#pragma unroll
    for (int t = 0; t < 16; ++t) {
      v16h bv = frag_b(&vbuf[(size_t)(t * 16 + n) * 32 + kbB]);
      o[t] = wmma_f16(pa, bv, o[t]);
    }
    __syncthreads();                    // all waves done before tiles overwritten
  }

#pragma unroll
  for (int r = 0; r < 8; ++r) lsum[r] = 1.f / lsum[r];
#pragma unroll
  for (int t = 0; t < 16; ++t) {
#pragma unroll
    for (int r = 0; r < 8; ++r)
      ot[((size_t)b * L + i0 + r + 8 * hi) * C + t * 16 + n] =
          (_Float16)(o[t][r] * lsum[r]);
  }
}

// ---------------------------------------------------------------------------
// Kernel 4: out = x + Wp*O + bp. One wave = one 16x16 tile; contiguous f32 I/O.
// ---------------------------------------------------------------------------
__global__ void proj_kernel(const _Float16* __restrict__ ot, const float* __restrict__ wp,
                            const float* __restrict__ bp, const float* __restrict__ x,
                            float* __restrict__ out) {
  const int wave = threadIdx.x >> 5, lane = threadIdx.x & 31;
  const int b = blockIdx.y;
  const int tt = blockIdx.x * 8 + wave;
  const int i0 = (tt >> 4) * 16, o0 = (tt & 15) * 16;
  const int n = lane & 15, hi = lane >> 4;
  const int kbA = hi * 16, kbB = hi * 16;

  const _Float16* arow = ot + ((size_t)b * L + i0 + n) * C + kbA;
  const float*    brow = wp + (size_t)(o0 + n) * C + kbB;

  v8f acc = {};
#pragma unroll
  for (int cb = 0; cb < 8; ++cb)
    acc = wmma_f16(frag_a(arow + cb * 32), frag_bf(brow + cb * 32), acc);

  const float bb = bp[o0 + n];
  const size_t base = ((size_t)b * C + o0 + n) * L + i0 + 8 * hi;
  const float* xr = x + base;
  float* orow = out + base;
#pragma unroll
  for (int r = 0; r < 8; ++r) orow[r] = xr[r] + acc[r] + bb;
}

// ---------------------------------------------------------------------------
extern "C" void kernel_launch(void* const* d_in, const int* in_sizes, int n_in,
                              void* d_out, int out_size, void* d_ws, size_t ws_size,
                              hipStream_t stream) {
  (void)in_sizes; (void)n_in; (void)out_size; (void)ws_size;
  const float* x    = (const float*)d_in[0];
  const float* gn_w = (const float*)d_in[1];
  const float* gn_b = (const float*)d_in[2];
  const float* wq   = (const float*)d_in[3];
  const float* bq   = (const float*)d_in[4];
  const float* wk   = (const float*)d_in[5];
  const float* bk   = (const float*)d_in[6];
  const float* wv   = (const float*)d_in[7];
  const float* bv   = (const float*)d_in[8];
  const float* wp   = (const float*)d_in[9];
  const float* bp   = (const float*)d_in[10];
  float* out = (float*)d_out;

  char* ws = (char*)d_ws;
  const size_t MAT = (size_t)B * L * C * sizeof(_Float16);   // 16 MB each
  _Float16* ht = (_Float16*)(ws);
  _Float16* qt = (_Float16*)(ws + MAT);
  _Float16* kt = (_Float16*)(ws + 2 * MAT);
  _Float16* vv = (_Float16*)(ws + 3 * MAT);
  _Float16* ot = (_Float16*)(ws + 4 * MAT);

  gn_kernel  <<<dim3(G, B),      256, 0, stream>>>(x, gn_w, gn_b, ht);
  qkv_kernel <<<dim3(512, B, 3), 256, 0, stream>>>(ht, wq, bq, wk, bk, wv, bv, qt, kt, vv);
  attn_kernel<<<dim3(256),       256, 0, stream>>>(qt, kt, vv, ot);
  proj_kernel<<<dim3(512, B),    256, 0, stream>>>(ot, wp, bp, x, out);
}
